// CRPEAttention_420906795537
// MI455X (gfx1250) — matrix operand order repaired and checked by reading, not compiled, not run on hardware
//
#include <hip/hip_runtime.h>
#include <hip/hip_bf16.h>

#define N_PTS   8192
#define C_IN    512
#define H_HEADS 8
#define D_DIM   64
#define QKV_C   1536          // 3*H*D
#define M_PAIRS 131072
#define BINS    48
#define AB      144           // 3*BINS
#define ABH     1152          // AB*H

typedef __attribute__((ext_vector_type(16))) __bf16 v16bf;
typedef __attribute__((ext_vector_type(8)))  float  v8f;
typedef __attribute__((ext_vector_type(4)))  unsigned int tdm_v4u;
typedef __attribute__((ext_vector_type(8)))  int          tdm_v8i;
typedef __attribute__((ext_vector_type(4)))  int          tdm_v4i;

union BFrag { v16bf v; uint4 q[2]; };

__device__ __forceinline__ unsigned short f2bf(float x) {
    unsigned int u = __float_as_uint(x);
    unsigned int r = u + 0x7FFFu + ((u >> 16) & 1u);   // round-to-nearest-even
    return (unsigned short)(r >> 16);
}

__device__ __forceinline__ void atomicMaxF32(float* addr, float v) {
    if (v >= 0.0f) atomicMax((int*)addr, __float_as_int(v));
    else           atomicMin((unsigned int*)addr, __float_as_uint(v));
}

// ---------------- conversion kernels ----------------
__global__ void conv_x_kernel(const float* __restrict__ xF, unsigned short* __restrict__ x16) {
    int t = blockIdx.x * blockDim.x + threadIdx.x;
    if (t < N_PTS * C_IN) x16[t] = f2bf(xF[t]);
}

// Wt[n][k] = W[k][n]  (K-contiguous bf16 for B fragments)
__global__ void conv_w_kernel(const float* __restrict__ W, unsigned short* __restrict__ Wt) {
    int t = blockIdx.x * blockDim.x + threadIdx.x;
    if (t >= QKV_C * C_IN) return;
    int n = t / C_IN, k = t % C_IN;
    Wt[t] = f2bf(W[(size_t)k * QKV_C + n]);
}

// tables (3,48,H,D) -> bf16 [h][ab][d] (K-contiguous over d)
__global__ void conv_tab_kernel(const float* __restrict__ qt, const float* __restrict__ kt,
                                unsigned short* __restrict__ qt16, unsigned short* __restrict__ kt16) {
    int t = blockIdx.x * blockDim.x + threadIdx.x;
    if (t >= 2 * H_HEADS * AB * D_DIM) return;
    int sel = t / (H_HEADS * AB * D_DIM);
    int r   = t % (H_HEADS * AB * D_DIM);
    int h   = r / (AB * D_DIM);
    int r2  = r % (AB * D_DIM);
    int ab  = r2 / D_DIM;
    int d   = r2 % D_DIM;
    const float* src = sel ? kt : qt;
    float val = src[((size_t)ab * H_HEADS + h) * D_DIM + d];
    (sel ? kt16 : qt16)[r] = f2bf(val);
}

// ---------------- WMMA GEMM: qkv = xF16 @ Wt16^T + b ----------------
// Block = 8 waves = 128 rows x 64 cols; B panel (64 cols x 512 K, bf16) staged
// into LDS by the Tensor Data Mover; each wave: 4 accumulators (16x64).
__global__ void __launch_bounds__(256)
gemm_qkv_wmma(const unsigned short* __restrict__ A,   // [N][512] bf16
              const unsigned short* __restrict__ Bt,  // [1536][512] bf16 (K-contig)
              const float* __restrict__ bias,
              float* __restrict__ qkv,                // [N][1536]
              unsigned short* __restrict__ q16,       // [N][512] bf16 (scaled)
              unsigned short* __restrict__ k16) {     // [N][512] bf16
    __shared__ __align__(128) unsigned short Bs[64 * C_IN];   // 64 KB

    const int lane = threadIdx.x & 31;
    const int wid  = threadIdx.x >> 5;                // 0..7
    const int bm   = blockIdx.x / (QKV_C / 64);       // 0..63   (row block of 128)
    const int bn   = blockIdx.x % (QKV_C / 64);       // 0..23   (col block of 64)

    // --- TDM stage: Wt[bn*64 .. +63][0..511] -> LDS (2D tile, 2B elems) ---
    if (wid == 0) {
        const unsigned short* src = Bt + (size_t)bn * 64 * C_IN;
        unsigned long long ga = (unsigned long long)src;
        unsigned int lds = (unsigned int)(unsigned long long)&Bs[0];
        tdm_v4u g0 = { 1u,                                   // count=1 valid descriptor
                       lds,                                  // D#.lds_addr
                       (unsigned int)ga,                     // global_addr[31:0]
                       (unsigned int)((ga >> 32) & 0x01FFFFFFu) | (2u << 30) }; // addr[56:32] | type=2
        tdm_v8i g1 = { 0x00010000,                           // data_size=1 (2 bytes)
                       (int)(512u << 16),                    // tensor_dim0 = 512 (bits 79:48)
                       (int)(1536u << 16),                   // tensor_dim1 = 1536 (bits 111:80)
                       (int)(512u << 16),                    // tile_dim0 = 512 (bits 127:112)
                       64,                                   // tile_dim1 = 64
                       512,                                  // tensor_dim0_stride = 512
                       0, 0 };
        tdm_v4i z = { 0, 0, 0, 0 };
#if __has_builtin(__builtin_amdgcn_tensor_load_to_lds)
#if __clang_major__ >= 23
        tdm_v8i z8 = { 0, 0, 0, 0, 0, 0, 0, 0 };
        __builtin_amdgcn_tensor_load_to_lds(g0, g1, z, z, z8, 0);
#else
        __builtin_amdgcn_tensor_load_to_lds(g0, g1, z, z, 0);
#endif
        __builtin_amdgcn_s_wait_tensorcnt(0);
#else
        for (int t = lane; t < 64 * C_IN / 8; t += 32)
            ((uint4*)Bs)[t] = ((const uint4*)src)[t];
#endif
    }
    __syncthreads();

    const int rowA     = bm * 128 + wid * 16 + (lane & 15);
    const int colLocal = lane & 15;
    const int koffA    = (lane >> 4) * 16;   // A: half-wave K split
    const int koffB    = (lane >> 4) * 16;   // B: half-wave K split

    v8f acc[4] = {{}, {}, {}, {}};
    const unsigned short* arow = A + (size_t)rowA * C_IN;
    for (int kb = 0; kb < C_IN; kb += 32) {
        BFrag a;
        a.q[0] = *(const uint4*)(arow + kb + koffA);
        a.q[1] = *(const uint4*)(arow + kb + koffA + 8);
        #pragma unroll
        for (int ct = 0; ct < 4; ++ct) {
            BFrag b;
            const unsigned short* bp = &Bs[(size_t)(ct * 16 + colLocal) * C_IN + kb + koffB];
            b.q[0] = *(const uint4*)bp;
            b.q[1] = *(const uint4*)(bp + 8);
            acc[ct] = __builtin_amdgcn_wmma_f32_16x16x32_bf16(false, a.v, false, b.v,
                                                              (short)0, acc[ct], false, false);
        }
    }

    const int roff = (lane >> 4) * 8;
    #pragma unroll
    for (int ct = 0; ct < 4; ++ct) {
        const int coln = bn * 64 + ct * 16 + colLocal;
        const float bi = bias[coln];
        const float scale = (coln < 512) ? 0.125f : 1.0f;    // D^-0.5 folded into q
        #pragma unroll
        for (int r = 0; r < 8; ++r) {
            int row = bm * 128 + wid * 16 + roff + r;
            float val = (acc[ct][r] + bi) * scale;
            qkv[(size_t)row * QKV_C + coln] = val;
            if (coln < 512)       q16[(size_t)row * 512 + coln]       = f2bf(val);
            else if (coln < 1024) k16[(size_t)row * 512 + coln - 512] = f2bf(val);
        }
    }
}

// ---------------- WMMA GEMM: dot[n][ab][h] = q[n,h,:] . tab[h,ab,:] ----------------
__global__ void gemm_dot_wmma(const unsigned short* __restrict__ A,   // [N][512] bf16
                              const unsigned short* __restrict__ T,   // [h][144][64] bf16
                              float* __restrict__ dot) {              // [N][1152] ([n][ab][h])
    const int lane = threadIdx.x & 31;
    const int w    = blockIdx.x * (blockDim.x >> 5) + (threadIdx.x >> 5);
    const int MT = N_PTS / 16, NT = AB / 16;            // 512 * 9
    if (w >= H_HEADS * MT * NT) return;
    const int h   = w / (MT * NT);
    const int rem = w % (MT * NT);
    const int mt  = rem / NT, nt = rem % NT;

    const int rowA = mt * 16 + (lane & 15);
    const int colB = nt * 16 + (lane & 15);
    const int koffA = (lane >> 4) * 16;
    const int koffB = (lane >> 4) * 16;

    v8f acc = {};
    const unsigned short* arow = A + (size_t)rowA * 512 + h * D_DIM;
    const unsigned short* bcol = T + ((size_t)h * AB + colB) * D_DIM;
    #pragma unroll
    for (int kb = 0; kb < D_DIM; kb += 32) {
        BFrag a, b;
        a.q[0] = *(const uint4*)(arow + kb + koffA);
        a.q[1] = *(const uint4*)(arow + kb + koffA + 8);
        b.q[0] = *(const uint4*)(bcol + kb + koffB);
        b.q[1] = *(const uint4*)(bcol + kb + koffB + 8);
        acc = __builtin_amdgcn_wmma_f32_16x16x32_bf16(false, a.v, false, b.v,
                                                      (short)0, acc, false, false);
    }
    const int ab = nt * 16 + (lane & 15);
    const int roff = (lane >> 4) * 8;
    #pragma unroll
    for (int r = 0; r < 8; ++r) {
        int row = mt * 16 + roff + r;
        dot[(size_t)row * ABH + ab * H_HEADS + h] = acc[r];
    }
}

// ---------------- init ----------------
__global__ void init_kernel(float* __restrict__ out, float* __restrict__ segmax,
                            float* __restrict__ denom) {
    int t = blockIdx.x * blockDim.x + threadIdx.x;
    if (t < N_PTS * 512) out[t] = 0.0f;
    if (t < N_PTS * H_HEADS) { segmax[t] = -__builtin_inff(); denom[t] = 0.0f; }
}

__device__ __forceinline__ void get_bins(const float* __restrict__ xC,
                                         long long qi, long long ki, int ib[3]) {
    #pragma unroll
    for (int a = 0; a < 3; ++a) {
        float rel = (xC[qi * 3 + a] - xC[ki * 3 + a]) * 20.0f + 24.0f; // /0.05 + BINS/2
        int v = (int)rel;                                              // trunc, as astype(int32)
        ib[a] = v < 0 ? 0 : (v > BINS - 1 ? BINS - 1 : v);
    }
}

// ---------------- pair logits + segment max ----------------
__global__ void pair_logits_kernel(const float* __restrict__ qkv, const float* __restrict__ xC,
                                   const long long* __restrict__ pidx,
                                   const float* __restrict__ qdot, const float* __restrict__ kdot,
                                   float* __restrict__ logits, float* __restrict__ segmax) {
    int t = blockIdx.x * blockDim.x + threadIdx.x;
    if (t >= M_PAIRS * H_HEADS) return;
    int m = t >> 3, h = t & 7;
    long long qi = pidx[m], ki = pidx[M_PAIRS + m];
    int ib[3]; get_bins(xC, qi, ki, ib);

    const float4* qp = (const float4*)(qkv + qi * QKV_C + h * D_DIM);
    const float4* kp = (const float4*)(qkv + ki * QKV_C + 512 + h * D_DIM);
    float s = 0.0f;
    #pragma unroll
    for (int i = 0; i < 16; ++i) {
        float4 a = qp[i], b = kp[i];
        s += a.x * b.x + a.y * b.y + a.z * b.z + a.w * b.w;
    }
    #pragma unroll
    for (int a = 0; a < 3; ++a) {
        int ab = a * BINS + ib[a];
        s += qdot[qi * ABH + ab * H_HEADS + h] + kdot[ki * ABH + ab * H_HEADS + h];
    }
    logits[t] = s;
    atomicMaxF32(&segmax[qi * H_HEADS + h], s);
}

// ---------------- exp + segment sum ----------------
__global__ void pair_exp_kernel(const long long* __restrict__ pidx,
                                float* __restrict__ logits, const float* __restrict__ segmax,
                                float* __restrict__ denom) {
    int t = blockIdx.x * blockDim.x + threadIdx.x;
    if (t >= M_PAIRS * H_HEADS) return;
    int m = t >> 3, h = t & 7;
    long long qi = pidx[m];
    float e = __expf(logits[t] - segmax[qi * H_HEADS + h]);
    logits[t] = e;
    atomicAdd(&denom[qi * H_HEADS + h], e);
}

// ---------------- weighted scatter-aggregate ----------------
__global__ void aggregate_kernel(const float* __restrict__ qkv, const float* __restrict__ xC,
                                 const long long* __restrict__ pidx,
                                 const float* __restrict__ vtab,     // (3,48,H,D)
                                 const float* __restrict__ e, const float* __restrict__ denom,
                                 float* __restrict__ out) {
    int m = blockIdx.x;
    long long qi = pidx[m], ki = pidx[M_PAIRS + m];
    int ib[3]; get_bins(xC, qi, ki, ib);
    for (int j = threadIdx.x; j < 512; j += blockDim.x) {
        int h = j >> 6;
        float attn = e[(size_t)m * H_HEADS + h] / denom[qi * H_HEADS + h];
        float vs = qkv[ki * QKV_C + 1024 + j];
        #pragma unroll
        for (int a = 0; a < 3; ++a)
            vs += vtab[(size_t)(a * BINS + ib[a]) * 512 + j];
        atomicAdd(&out[qi * 512 + j], attn * vs);
    }
}

extern "C" void kernel_launch(void* const* d_in, const int* in_sizes, int n_in,
                              void* d_out, int out_size, void* d_ws, size_t ws_size,
                              hipStream_t stream) {
    const float*      x_F   = (const float*)d_in[0];
    const float*      x_C   = (const float*)d_in[1];
    const long long*  pidx  = (const long long*)d_in[2];
    const float*      W_qkv = (const float*)d_in[3];
    const float*      b_qkv = (const float*)d_in[4];
    const float*      q_tab = (const float*)d_in[5];
    const float*      k_tab = (const float*)d_in[6];
    const float*      v_tab = (const float*)d_in[7];
    float*            out   = (float*)d_out;

    char* w = (char*)d_ws;
    auto carve = [&](size_t bytes) { char* p = w; w += (bytes + 255) & ~(size_t)255; return p; };
    unsigned short* x16   = (unsigned short*)carve((size_t)N_PTS * C_IN * 2);
    unsigned short* Wt16  = (unsigned short*)carve((size_t)QKV_C * C_IN * 2);
    unsigned short* qt16  = (unsigned short*)carve((size_t)H_HEADS * AB * D_DIM * 2);
    unsigned short* kt16  = (unsigned short*)carve((size_t)H_HEADS * AB * D_DIM * 2);
    float*          qkv   = (float*)carve((size_t)N_PTS * QKV_C * 4);
    unsigned short* q16   = (unsigned short*)carve((size_t)N_PTS * 512 * 2);
    unsigned short* k16   = (unsigned short*)carve((size_t)N_PTS * 512 * 2);
    float*          qdot  = (float*)carve((size_t)N_PTS * ABH * 4);
    float*          kdot  = (float*)carve((size_t)N_PTS * ABH * 4);
    float*          logit = (float*)carve((size_t)M_PAIRS * H_HEADS * 4);
    float*          smax  = (float*)carve((size_t)N_PTS * H_HEADS * 4);
    float*          den   = (float*)carve((size_t)N_PTS * H_HEADS * 4);

    const int B = 256;
    conv_x_kernel<<<(N_PTS * C_IN + B - 1) / B, B, 0, stream>>>(x_F, x16);
    conv_w_kernel<<<(QKV_C * C_IN + B - 1) / B, B, 0, stream>>>(W_qkv, Wt16);
    conv_tab_kernel<<<(2 * H_HEADS * AB * D_DIM + B - 1) / B, B, 0, stream>>>(q_tab, k_tab, qt16, kt16);

    // 64 row-blocks (128 rows) x 24 col-blocks (64 cols) = 1536 blocks
    gemm_qkv_wmma<<<(N_PTS / 128) * (QKV_C / 64), B, 0, stream>>>(x16, Wt16, b_qkv, qkv, q16, k16);
    // 8*512*9 wave tiles, 8 waves/block
    gemm_dot_wmma<<<H_HEADS * (N_PTS / 16) * (AB / 16) / 8, B, 0, stream>>>(q16, qt16, qdot);
    gemm_dot_wmma<<<H_HEADS * (N_PTS / 16) * (AB / 16) / 8, B, 0, stream>>>(k16, kt16, kdot);

    init_kernel<<<(N_PTS * 512 + B - 1) / B, B, 0, stream>>>(out, smax, den);
    pair_logits_kernel<<<(M_PAIRS * H_HEADS + B - 1) / B, B, 0, stream>>>(qkv, x_C, pidx, qdot, kdot, logit, smax);
    pair_exp_kernel<<<(M_PAIRS * H_HEADS + B - 1) / B, B, 0, stream>>>(pidx, logit, smax, den);
    aggregate_kernel<<<M_PAIRS, B, 0, stream>>>(qkv, x_C, pidx, v_tab, logit, den, out);
}
